// Model_26096221290673
// MI455X (gfx1250) — matrix operand order repaired and checked by reading, not compiled, not run on hardware
//
#include <hip/hip_runtime.h>
#include <hip/hip_bf16.h>
#include <stdint.h>
#include <stddef.h>

// ---------------------------------------------------------------------------
// LSTM (T=2048, B=64, I=512, H=256) + per-step FC (O=128) for MI455X/gfx1250.
//  K1: convert X fp32 -> f16
//  K2: pack W_ih / W_hh / W_fc into WMMA B-operand lane layout (f16)
//  K3: big WMMA GEMM: preact = X @ W_ih^T + (b_ih+b_hh), stored tile-packed
//  K4: persistent single-WGP recurrence: 32 waves, h in LDS (f16), c in regs.
//      W_hh k-blocks 0..3 cached in LDS (256KB) -> low-latency B operands;
//      k-blocks 4..7 streamed from L2, interleaved so loads overlap LDS WMMAs.
//      288KB dynamic LDS total (<= 320KB/WGP).
//  K5: big WMMA GEMM: out = h_all @ W_fc^T + b_fc (B tiles fully hoisted)
// ---------------------------------------------------------------------------

typedef __attribute__((ext_vector_type(16))) _Float16 v16h;
typedef __attribute__((ext_vector_type(8)))  _Float16 v8h;
typedef __attribute__((ext_vector_type(4)))  _Float16 v4h;
typedef __attribute__((ext_vector_type(8)))  float    v8f;
typedef __attribute__((ext_vector_type(4)))  float    v4f;

#define T_STEPS 2048
#define BSZ     64
#define ISZ     512
#define HSZ     256
#define OSZ     128
#define GSZ     (4*HSZ)   // 1024 gate width

#define BLDS_HALFS (64 * 4 * 512)        // 64 nt * 4 kb * (32 lanes * 16) halfs
#define HBUF_HALFS (BSZ * HSZ)           // 16384 halfs
#define RECUR_LDS_BYTES ((BLDS_HALFS + HBUF_HALFS) * 2)   // 294912 = 288 KB

union F8 { v8f f; v4f q[2]; };
union A16 { v16h v; v8h h[2]; };

// ---- fast activations ------------------------------------------------------
__device__ __forceinline__ float sigf(float x) {
  return 1.0f / (1.0f + __expf(-x));
}
__device__ __forceinline__ float tanh_fast(float x) {
  x = fminf(fmaxf(x, -10.0f), 10.0f);
  float e = __expf(2.0f * x);
  return (e - 1.0f) / (e + 1.0f);
}

// ---- WMMA operand loaders --------------------------------------------------
// A tile (16x32 f16) from row-major source, row stride `stride` (halfs).
// Lane layout: lanes 0-15 -> rows 0-15, K = 0..7 & 16..23;
//              lanes 16-31 -> rows 0-15, K = 8..15 & 24..31.
__device__ __forceinline__ v16h load_a_tile(const _Float16* base, int stride) {
  int lane = threadIdx.x & 31;
  int m  = lane & 15;
  int hi = lane >> 4;
  const _Float16* p = base + m * stride + hi * 8;
  A16 u;
  u.h[0] = *(const v8h*)p;
  u.h[1] = *(const v8h*)(p + 16);
  return u.v;
}

// B tile (32x16 f16) from pre-packed layout [nt][kb][lane][16] (contiguous).
__device__ __forceinline__ v16h load_b_tile(const _Float16* packed, int nt,
                                            int kb, int kblks) {
  int lane = threadIdx.x & 31;
  const _Float16* p = packed + ((size_t)(nt * kblks + kb) * 32 + lane) * 16;
  return *(const v16h*)p;
}

// B tile from the LDS-resident W_hh panel (k-blocks 0..3).
__device__ __forceinline__ v16h load_b_tile_lds(const _Float16* Blds, int nt,
                                                int kb) {
  int lane = threadIdx.x & 31;
  const _Float16* p = Blds + (((nt * 4 + kb) * 32 + lane) * 16);
  return *(const v16h*)p;
}

// ---- K1: fp32 -> f16 conversion (4 elems/thread) ---------------------------
__global__ __launch_bounds__(256) void cvt_f32_to_f16(const float* __restrict__ src,
                                                      _Float16* __restrict__ dst,
                                                      int n4) {
  int i = blockIdx.x * blockDim.x + threadIdx.x;
  if (i >= n4) return;
  v4f x = *(const v4f*)(src + (size_t)i * 4);
  v4h y;
  y[0] = (_Float16)x[0]; y[1] = (_Float16)x[1];
  y[2] = (_Float16)x[2]; y[3] = (_Float16)x[3];
  *(v4h*)(dst + (size_t)i * 4) = y;
}

// ---- K2: pack row-major W[N][K] fp32 into WMMA B layout (f16) --------------
// dst[nt][kb][lane][i] = W[nt*16 + (lane&15)][kb*32 + (lane>>4)*8 + i + (i>=8 ? 8 : 0)]
__global__ __launch_bounds__(128) void pack_wmma_b(const float* __restrict__ W,
                                                   _Float16* __restrict__ dst,
                                                   int N, int K) {
  int tid  = blockIdx.x * blockDim.x + threadIdx.x;
  int lane = tid & 31;
  int idx  = tid >> 5;
  int kblks = K >> 5;
  int nt = idx / kblks;
  int kb = idx - nt * kblks;
  if (nt >= (N >> 4)) return;
  int n  = nt * 16 + (lane & 15);
  int k0 = kb * 32 + (lane >> 4) * 8;
  const float* src = W + (size_t)n * K + k0;
  _Float16* d = dst + (size_t)tid * 16;
#pragma unroll
  for (int i = 0; i < 16; ++i) d[i] = (_Float16)src[i + (i >= 8 ? 8 : 0)];
}

// ---- K3: input projection GEMM ---------------------------------------------
// preact (tile-packed, f32): [t][mt(4)][ht(16)][gate(4)][lane(32)][r(8)]
__global__ __launch_bounds__(256) void lstm_input_gemm(
    const _Float16* __restrict__ X16, const _Float16* __restrict__ WihP,
    const float* __restrict__ b_ih, const float* __restrict__ b_hh,
    float* __restrict__ preP) {
  const int KB = ISZ / 32;           // 16 k-blocks
  int strip = blockIdx.x;            // 16-row strip of M = T*B
  int w    = threadIdx.x >> 5;       // wave 0..7
  int lane = threadIdx.x & 31;

  v8f acc[8];
#pragma unroll
  for (int j = 0; j < 8; ++j) acc[j] = (v8f){0,0,0,0,0,0,0,0};

  const _Float16* Abase = X16 + (size_t)strip * 16 * ISZ;
  for (int kb = 0; kb < KB; ++kb) {
    v16h a = load_a_tile(Abase + kb * 32, ISZ);
    v16h b[8];
#pragma unroll
    for (int j = 0; j < 8; ++j) b[j] = load_b_tile(WihP, w + j * 8, kb, KB);
#pragma unroll
    for (int j = 0; j < 8; ++j)
      acc[j] = __builtin_amdgcn_wmma_f32_16x16x32_f16(
          false, a, false, b[j], (short)0, acc[j], false, false);
  }

  int t = strip >> 2, mt = strip & 3;
#pragma unroll
  for (int j = 0; j < 8; ++j) {
    int nt = w + j * 8;
    int g  = nt >> 4, ht = nt & 15;
    int n  = nt * 16 + (lane & 15);
    float bias = b_ih[n] + b_hh[n];
    F8 s;
#pragma unroll
    for (int r = 0; r < 8; ++r) s.f[r] = acc[j][r] + bias;
    float* dst = preP + ((((size_t)t * 4 + mt) * 16 + ht) * 4 + g) * 256
                      + (size_t)lane * 8;
    *(v4f*)dst       = s.q[0];
    *(v4f*)(dst + 4) = s.q[1];
  }
}

// ---- K4: persistent recurrence (one WGP, 32 waves) -------------------------
__device__ __forceinline__ void lstm_update(const v8f a[4],
                                            const float* __restrict__ pp,
                                            v8f& cst, _Float16* hbuf,
                                            _Float16* __restrict__ h_all,
                                            int t, int mrow_base, int col) {
  F8 pr[4];
#pragma unroll
  for (int g = 0; g < 4; ++g) {
    pr[g].q[0] = *(const v4f*)(pp + g * 256);
    pr[g].q[1] = *(const v4f*)(pp + g * 256 + 4);
  }
#pragma unroll
  for (int r = 0; r < 8; ++r) {
    float iv = sigf(a[0][r] + pr[0].f[r]);
    float fv = sigf(a[1][r] + pr[1].f[r]);
    float gv = tanh_fast(a[2][r] + pr[2].f[r]);
    float ov = sigf(a[3][r] + pr[3].f[r]);
    float cc = fv * cst[r] + iv * gv;
    cst[r] = cc;
    float hv = ov * tanh_fast(cc);
    int m = mrow_base + r;
    hbuf[m * HSZ + col] = (_Float16)hv;
    h_all[((size_t)t * BSZ + m) * HSZ + col] = (_Float16)hv;
  }
}

__global__ __launch_bounds__(1024, 1) void lstm_recurrence(
    const float* __restrict__ preP, const _Float16* __restrict__ WhhP,
    _Float16* __restrict__ h_all) {
  extern __shared__ _Float16 smem[];           // 288 KB dynamic LDS
  _Float16* Blds = smem;                       // 256 KB: W_hh kb 0..3
  _Float16* hbuf = smem + BLDS_HALFS;          //  32 KB: h_t (f16)

  const int KB = HSZ / 32;                     // 8 k-blocks
  int tid  = threadIdx.x;
  int w    = tid >> 5;                         // wave 0..31
  int lane = tid & 31;

  // one-time prefill of the LDS W_hh panel (kb 0..3 of each nt), 16B chunks.
  // LDS tile index t4 = nt*4 + kb  <->  global tile index nt*8 + kb.
  for (int c = tid; c < BLDS_HALFS / 8; c += 1024) {
    int t4 = c >> 6;                 // tile (512 halfs = 64 chunks of 8)
    int within = c & 63;
    int nt = t4 >> 2, kb = t4 & 3;
    *(v8h*)(Blds + (size_t)t4 * 512 + within * 8) =
        *(const v8h*)(WhhP + (size_t)(nt * 8 + kb) * 512 + within * 8);
  }
  // zero h_0
  for (int i = tid; i < HBUF_HALFS / 2; i += 1024)
    ((uint32_t*)hbuf)[i] = 0u;
  __syncthreads();

  // wave w owns M-tile mt and two 16-column groups ht0 / ht1 (all 4 gates)
  int mt  = w & 3;
  int ht0 = w >> 2;        // 0..7
  int ht1 = ht0 + 8;       // 8..15
  int hi  = lane >> 4;
  int mrow_base = mt * 16 + 8 * hi;      // + r gives batch row
  int col0 = ht0 * 16 + (lane & 15);
  int col1 = ht1 * 16 + (lane & 15);

  v8f c0 = (v8f){0,0,0,0,0,0,0,0};
  v8f c1 = (v8f){0,0,0,0,0,0,0,0};

  const size_t preStepStride = (size_t)4 * 16 * 4 * 256;  // floats per step
  const float* pp0 = preP + (((size_t)mt * 16 + ht0) * 4) * 256
                          + (size_t)lane * 8;
  const float* pp1 = preP + (((size_t)mt * 16 + ht1) * 4) * 256
                          + (size_t)lane * 8;

  // interleave global k-blocks (4..7) with LDS k-blocks (0..3) so the L2
  // loads overlap the low-latency LDS WMMA bursts.
  const int kbOrder[8] = {4, 0, 5, 1, 6, 2, 7, 3};

  for (int t = 0; t < T_STEPS; ++t) {
    // prefetch next step's pre-activations into cache (global_prefetch_b8)
    if (t + 1 < T_STEPS) {
      __builtin_prefetch(pp0 + preStepStride, 0, 0);
      __builtin_prefetch(pp1 + preStepStride, 0, 0);
    }

    v8f a0[4], a1[4];
#pragma unroll
    for (int g = 0; g < 4; ++g) {
      a0[g] = (v8f){0,0,0,0,0,0,0,0};
      a1[g] = (v8f){0,0,0,0,0,0,0,0};
    }
    const _Float16* Ab = hbuf + mt * 16 * HSZ;
#pragma unroll
    for (int ki = 0; ki < KB; ++ki) {
      int kb = kbOrder[ki];
      v16h a = load_a_tile(Ab + kb * 32, HSZ);   // shared A for both groups
      v16h b0[4], b1[4];
      if (kb < 4) {
#pragma unroll
        for (int g = 0; g < 4; ++g) {
          b0[g] = load_b_tile_lds(Blds, g * 16 + ht0, kb);
          b1[g] = load_b_tile_lds(Blds, g * 16 + ht1, kb);
        }
      } else {
#pragma unroll
        for (int g = 0; g < 4; ++g) {
          b0[g] = load_b_tile(WhhP, g * 16 + ht0, kb, KB);
          b1[g] = load_b_tile(WhhP, g * 16 + ht1, kb, KB);
        }
      }
#pragma unroll
      for (int g = 0; g < 4; ++g) {
        a0[g] = __builtin_amdgcn_wmma_f32_16x16x32_f16(
            false, a, false, b0[g], (short)0, a0[g], false, false);
        a1[g] = __builtin_amdgcn_wmma_f32_16x16x32_f16(
            false, a, false, b1[g], (short)0, a1[g], false, false);
      }
    }
    __syncthreads();   // all hbuf reads for step t complete

    lstm_update(a0, pp0, c0, hbuf, h_all, t, mrow_base, col0);
    lstm_update(a1, pp1, c1, hbuf, h_all, t, mrow_base, col1);
    pp0 += preStepStride;
    pp1 += preStepStride;

    __syncthreads();   // h_{t+1} visible before next step's WMMA reads
  }
}

// ---- K5: output FC GEMM ----------------------------------------------------
__global__ __launch_bounds__(256) void fc_gemm(const _Float16* __restrict__ h_all,
                                               const _Float16* __restrict__ WfcP,
                                               const float* __restrict__ b_fc,
                                               float* __restrict__ out) {
  const int KB = HSZ / 32;               // 8 k-blocks
  int strip = blockIdx.x;                // 16-row strip of M = T*B
  int w    = threadIdx.x >> 5;           // wave -> N-tile (8 tiles, N=128)
  int lane = threadIdx.x & 31;

  // hoist the whole B panel for this wave's N-tile (8 tiles = 64 VGPRs)
  v16h b[8];
#pragma unroll
  for (int kb = 0; kb < KB; ++kb) b[kb] = load_b_tile(WfcP, w, kb, KB);

  v8f acc = (v8f){0,0,0,0,0,0,0,0};
  const _Float16* Ab = h_all + (size_t)strip * 16 * HSZ;
#pragma unroll
  for (int kb = 0; kb < KB; ++kb) {
    v16h a = load_a_tile(Ab + kb * 32, HSZ);
    acc = __builtin_amdgcn_wmma_f32_16x16x32_f16(
        false, a, false, b[kb], (short)0, acc, false, false);
  }
  int n  = w * 16 + (lane & 15);
  float bias = b_fc[n];
  int hi = lane >> 4;
#pragma unroll
  for (int r = 0; r < 8; ++r)
    out[((size_t)strip * 16 + r + 8 * hi) * OSZ + n] = acc[r] + bias;
}

// ---------------------------------------------------------------------------
extern "C" void kernel_launch(void* const* d_in, const int* in_sizes, int n_in,
                              void* d_out, int out_size, void* d_ws, size_t ws_size,
                              hipStream_t stream) {
  (void)in_sizes; (void)n_in; (void)out_size; (void)ws_size;
  const float* X   = (const float*)d_in[0];   // [T,B,I]
  const float* Wih = (const float*)d_in[1];   // [4H,I]
  const float* Whh = (const float*)d_in[2];   // [4H,H]
  const float* bih = (const float*)d_in[3];   // [4H]
  const float* bhh = (const float*)d_in[4];   // [4H]
  const float* Wfc = (const float*)d_in[5];   // [O,H]
  const float* bfc = (const float*)d_in[6];   // [O]
  float* out = (float*)d_out;                 // [T,B,O]

  // workspace layout (bytes)
  char* ws = (char*)d_ws;
  _Float16* X16  = (_Float16*)(ws);                         // 134,217,728
  _Float16* WihP = (_Float16*)(ws + 134217728ull);          //   1,048,576
  _Float16* WhhP = (_Float16*)(ws + 135266304ull);          //     524,288
  _Float16* WfcP = (_Float16*)(ws + 135790592ull);          //      65,536
  float*    preP = (float*)   (ws + 135856128ull);          // 536,870,912
  _Float16* hAll = (_Float16*)(ws + 672727040ull);          //  67,108,864

  // K1: X -> f16  (67,108,864 elems, 4/thread)
  cvt_f32_to_f16<<<65536, 256, 0, stream>>>(X, X16, 16777216);
  // K2: weight packing
  pack_wmma_b<<<256, 128, 0, stream>>>(Wih, WihP, GSZ, ISZ);  // 64*16*32 thr
  pack_wmma_b<<<128, 128, 0, stream>>>(Whh, WhhP, GSZ, HSZ);  // 64*8*32 thr
  pack_wmma_b<<<16,  128, 0, stream>>>(Wfc, WfcP, OSZ, HSZ);  // 8*8*32 thr
  // K3: input projection (8192 strips of 16 rows)
  lstm_input_gemm<<<8192, 256, 0, stream>>>(X16, WihP, bih, bhh, preP);
  // K4: sequential recurrence — one persistent workgroup (32 waves),
  //     288 KB dynamic LDS (W_hh kb0..3 panel + h buffer)
  lstm_recurrence<<<1, 1024, RECUR_LDS_BYTES, stream>>>(preP, WhhP, hAll);
  // K5: output FC
  fc_gemm<<<8192, 256, 0, stream>>>(hAll, WfcP, bfc, out);
}
